// Kernel_10494947694770001583_53472342835835
// MI455X (gfx1250) — matrix operand order repaired and checked by reading, not compiled, not run on hardware
//
#include <hip/hip_runtime.h>
#include <float.h>

typedef __attribute__((ext_vector_type(16))) __bf16 v16bf;
typedef __attribute__((ext_vector_type(8)))  float  v8f;

constexpr int kN  = 32;
constexpr int kC  = 256;
constexpr int kH  = 56;
constexpr int kW  = 56;
constexpr int kHW = 56 * 56;     // 3136
constexpr int kHP = 68;          // padded H (halo 6 each side)
constexpr int kWP = 68;          // padded W

__device__ __forceinline__ unsigned short f2bf(float f) {
  unsigned int u = __float_as_uint(f);
  u += 0x7FFFu + ((u >> 16) & 1u);            // round-to-nearest-even
  return (unsigned short)(u >> 16);
}

union FragBF { uint4 u4[2]; v16bf v; };

// ---- zero-fill padded bf16 x buffer (halo stays zero) ----------------------
__global__ void zero_xb_kernel(uint4* __restrict__ p) {
  p[(size_t)blockIdx.x * 256 + threadIdx.x] = make_uint4(0u, 0u, 0u, 0u);
}

// ---- x (NCHW f32) -> xb (N,68,68,C) bf16 channel-last, via LDS transpose ---
__global__ __launch_bounds__(256)
void pack_x_kernel(const float* __restrict__ x, unsigned short* __restrict__ xb) {
  __shared__ unsigned short tile[64][57];
  const int h = blockIdx.x, n = blockIdx.y, t = threadIdx.x;
  for (int cb = 0; cb < 4; ++cb) {
    // load 64 ci x 56 w, coalesced over w
#pragma unroll
    for (int i = 0; i < 14; ++i) {
      int idx = i * 256 + t;                  // 3584 = 14*256
      int ci = idx / 56, w = idx % 56;
      tile[ci][w] = f2bf(x[((size_t)(n * kC + cb * 64 + ci) * kH + h) * kW + w]);
    }
    __syncthreads();
    // store 56 w x 64 ci, coalesced over ci (128B runs)
#pragma unroll
    for (int i = 0; i < 14; ++i) {
      int w  = i * 4 + (t >> 6);
      int ci = t & 63;
      xb[((size_t)((n * kHP + h + 6) * kWP) + (w + 6)) * 256 + cb * 64 + ci] =
          tile[ci][w];
    }
    __syncthreads();
  }
}

// ---- w7 (f32 [co][ci*25+tap]) -> bf16 [tap][cc][co][32] --------------------
__global__ void pack_w_kernel(const float* __restrict__ w7,
                              unsigned short* __restrict__ wp) {
  int idx = blockIdx.x * 256 + threadIdx.x;   // 25*8*256*32 = 1,638,400
  int tap = idx >> 16;
  int r   = idx & 65535;
  int cc  = r >> 13;
  int r2  = r & 8191;
  int co  = r2 >> 5;
  int kk  = r2 & 31;
  wp[idx] = f2bf(w7[co * 6400 + (cc * 32 + kk) * 25 + tap]);
}

// ---- t1 = max over W : one wave per (n,c,h) --------------------------------
__global__ void rowmax_kernel(const float* __restrict__ x,
                              float* __restrict__ t1) {
  int wv = threadIdx.x >> 5, lane = threadIdx.x & 31;
  int row = blockIdx.x * 8 + wv;
  const float* p = x + (size_t)row * kW;
  float v = p[lane];
  if (lane < kW - 32) v = fmaxf(v, p[lane + 32]);
  for (int off = 16; off; off >>= 1) v = fmaxf(v, __shfl_xor(v, off, 32));
  if (lane == 0) t1[row] = v;
}

// ---- t6[n,h,w] = sum_c tanh(relu(x)) * w6[c] -------------------------------
__global__ void t6_kernel(const float* __restrict__ x,
                          const float* __restrict__ w6,
                          float* __restrict__ t6) {
  int h = blockIdx.x, n = blockIdx.y, t = threadIdx.x;
  if (t >= kW) return;
  float acc = 0.f;
  size_t base = (size_t)n * kC * kHW + (size_t)h * kW + t;
  for (int c = 0; c < kC; ++c)
    acc += tanhf(fmaxf(x[base + (size_t)c * kHW], 0.f)) * w6[c];
  t6[(n * kH + h) * kW + t] = acc;
}

// ---- t9 = softmax over W of t6 ---------------------------------------------
__global__ void softmax_kernel(const float* __restrict__ t6,
                               float* __restrict__ t9) {
  int wv = threadIdx.x >> 5, lane = threadIdx.x & 31;
  int row = blockIdx.x * 8 + wv;
  const float* p = t6 + (size_t)row * kW;
  float v0 = p[lane];
  float v1 = (lane < kW - 32) ? p[lane + 32] : -FLT_MAX;
  float m = fmaxf(v0, v1);
  for (int off = 16; off; off >>= 1) m = fmaxf(m, __shfl_xor(m, off, 32));
  float e0 = __expf(v0 - m);
  float e1 = (lane < kW - 32) ? __expf(v1 - m) : 0.f;
  float s = e0 + e1;
  for (int off = 16; off; off >>= 1) s += __shfl_xor(s, off, 32);
  float inv = 1.f / s;
  t9[row * kW + lane] = e0 * inv;
  if (lane < kW - 32) t9[row * kW + 32 + lane] = e1 * inv;
}

// ---- dilated 5x5 conv: LDS-free implicit GEMM, 64x64 wave tiles ------------
// Block = 4 waves (128 thr), block tile = 256 couts x 64 pixels (8x8 patch).
// Each wave: 64 couts x 64 pixels = 4x4 WMMA tiles, K = 25 taps x 256 ci.
// A and B fragments are direct aligned b128 global loads (L0/L2 resident).
__global__ __launch_bounds__(128)
void conv_wmma_kernel(const unsigned short* __restrict__ xb,
                      const unsigned short* __restrict__ wp,
                      const float* __restrict__ x,
                      const float* __restrict__ t1,
                      const float* __restrict__ t9,
                      float* __restrict__ out) {
  const int tid  = threadIdx.x;
  const int lane = tid & 31;
  const int wv   = tid >> 5;                  // cout sub-block, x64
  const int l16  = lane & 15;
  const int half = lane >> 4;
  const int n  = blockIdx.z;
  const int h0 = blockIdx.y * 8;
  const int w0 = blockIdx.x * 8;
  const int co0 = wv * 64;
  const int abase = half ? 8 : 0;             // A frag K-base (ISA layout)
  const int kb    = half ? 8 : 0;             // B frag K-base (ISA layout)
  const int phl = l16 >> 3, pwl = l16 & 7;    // pixel coords within 8x8 patch

  v8f acc[4][4] = {};

  for (int tap = 0; tap < 25; ++tap) {
    const int dh = (tap / 5) * 3 - 6;
    const int dw = (tap % 5) * 3 - 6;

    if (tap + 1 < 25)                         // prefetch next tap's weights
      __builtin_prefetch(wp + (size_t)(tap + 1) * 65536 + tid * 512, 0, 1);

    const unsigned short* pA[4];
#pragma unroll
    for (int i = 0; i < 4; ++i)
      pA[i] = wp + ((size_t)tap * 2048 + (co0 + i * 16 + l16)) * 32 + abase;

    const unsigned short* pB[4];
#pragma unroll
    for (int j = 0; j < 4; ++j) {
      int hp  = h0 + 6 + dh + j * 2 + phl;
      int wpx = w0 + 6 + dw + pwl;
      pB[j] = xb + ((size_t)((n * kHP + hp) * kWP + wpx)) * 256 + kb;
    }

#pragma unroll
    for (int cc = 0; cc < 8; ++cc) {
      FragBF fa[4], fb[4];
#pragma unroll
      for (int i = 0; i < 4; ++i) {           // A: M=cout, K=ci (64B rows)
        fa[i].u4[0] = *(const uint4*)(pA[i] + cc * 8192);
        fa[i].u4[1] = *(const uint4*)(pA[i] + cc * 8192 + 16);
      }
#pragma unroll
      for (int j = 0; j < 4; ++j) {           // B: K=ci, N=pixel
        fb[j].u4[0] = *(const uint4*)(pB[j] + cc * 32);
        fb[j].u4[1] = *(const uint4*)(pB[j] + cc * 32 + 16);
      }
#pragma unroll
      for (int i = 0; i < 4; ++i)
#pragma unroll
        for (int j = 0; j < 4; ++j)
          acc[i][j] = __builtin_amdgcn_wmma_f32_16x16x32_bf16(
              false, fa[i].v, false, fb[j].v, (short)0, acc[i][j],
              false, false);
    }
  }

  // Fused epilogue: out = t1 - (t9 * roll(x,2,H) + x * t7)
#pragma unroll
  for (int j = 0; j < 4; ++j) {
    int h = h0 + j * 2 + phl;
    int w = w0 + pwl;
    int hm2 = (h + kH - 2) % kH;
    float t9v = t9[(n * kH + h) * kW + w];
#pragma unroll
    for (int i = 0; i < 4; ++i) {
#pragma unroll
      for (int r = 0; r < 8; ++r) {
        int co = co0 + i * 16 + r + half * 8;
        size_t base = ((size_t)(n * kC + co)) * kHW;
        float xv  = x[base + h * kW + w];
        float x5  = x[base + hm2 * kW + w];
        float t1v = t1[(n * kC + co) * kH + h];
        out[base + h * kW + w] = t1v - (t9v * x5 + xv * acc[i][j][r]);
      }
    }
  }
}

extern "C" void kernel_launch(void* const* d_in, const int* in_sizes, int n_in,
                              void* d_out, int out_size, void* d_ws, size_t ws_size,
                              hipStream_t stream) {
  const float* x  = (const float*)d_in[0];    // (32,256,56,56) f32
  const float* w6 = (const float*)d_in[1];    // (256,1) f32
  const float* w7 = (const float*)d_in[2];    // (256,6400) f32
  float* out = (float*)d_out;

  char* ws = (char*)d_ws;
  // xb: 32*68*68*256 bf16 = 75,759,616 B
  unsigned short* xb = (unsigned short*)ws;
  unsigned short* wpk = (unsigned short*)(ws + 75759616);        // 3,276,800 B
  float* t1b = (float*)(ws + 75759616 + 3276800);                // 1,835,008 B
  float* t6b = (float*)(ws + 75759616 + 3276800 + 1835008);      //   401,408 B
  float* t9b = (float*)(ws + 75759616 + 3276800 + 1835008 + 401408);

  zero_xb_kernel <<<18496, 256, 0, stream>>>((uint4*)xb);        // 4,734,976 uint4
  pack_x_kernel  <<<dim3(kH, kN), 256, 0, stream>>>(x, xb);
  pack_w_kernel  <<<6400, 256, 0, stream>>>(w7, wpk);
  rowmax_kernel  <<<(kN * kC * kH) / 8, 256, 0, stream>>>(x, t1b);
  t6_kernel      <<<dim3(kH, kN), 64, 0, stream>>>(x, w6, t6b);
  softmax_kernel <<<(kN * kH) / 8, 256, 0, stream>>>(t6b, t9b);
  conv_wmma_kernel<<<dim3(7, 7, kN), 128, 0, stream>>>(xb, wpk, x, t1b, t9b, out);
}
